// Attention_7799660610022
// MI455X (gfx1250) — compile-verified, matrix operand order checked by reading, not run on hardware
//
#include <hip/hip_runtime.h>

// ---------------------------------------------------------------------------
// MLA attention forward for gfx1250 (MI455X).
//  - All matmuls via v_wmma_f32_16x16x32_bf16 (fp32 -> bf16, fp32 accumulate)
//  - GEMM stages 64x32 tiles through LDS with gfx1250 async global->LDS
//    copies (global_load_async_to_lds_b128 / s_wait_asynccnt), double-buffered
//  - Fused attention keeps the whole 16x2048 prob row-block in LDS (128 KB of
//    the WGP's 320 KB)
// Workspace requirement: ~122 MB (see offsets below).
// ---------------------------------------------------------------------------

typedef __attribute__((ext_vector_type(16))) __bf16 v16bf;
typedef __attribute__((ext_vector_type(8)))  float  v8f;

constexpr int S_     = 2048;
constexpr int H_     = 5120;
constexpr int NH_    = 16;
constexpr int QLORA_ = 1536;
constexpr int KVLORA_= 512;
constexpr int NOPE_  = 128;
constexpr int ROPE_  = 64;
constexpr int QK_    = 192;   // NOPE + ROPE
constexpr int VD_    = 128;
constexpr float EPS_ = 1e-6f;

// Load 16 consecutive fp32, convert to bf16 fragment half-row.
// Fragment layout (A and "B from row-major N x K weights"):
//   lane L: row = base_row + (L & 15), k offset = ((L >> 4) << 4) + j
static __device__ inline v16bf frag_row(const float* __restrict__ p) {
  v16bf r;
#pragma unroll
  for (int j = 0; j < 16; ++j) r[j] = (__bf16)p[j];
  return r;
}

// B fragment from a K x N row-major matrix (e.g. V): lane L holds column
// n = (L & 15); k rows = kbase + ((L >> 4) << 4) + j, stride = row pitch.
static __device__ inline v16bf frag_bT(const float* __restrict__ base, int stride) {
  const int lane = threadIdx.x & 31;
  const float* p = base + (size_t)(((lane >> 4) << 4)) * stride + (lane & 15);
  v16bf r;
#pragma unroll
  for (int j = 0; j < 16; ++j) r[j] = (__bf16)p[(size_t)j * stride];
  return r;
}

// Async-copy a 64-row x 32-float tile (row stride `ld` floats in global,
// packed 32-float rows in LDS) using gfx1250 async global->LDS b128 loads.
// 512 16-byte chunks split over 128 threads.  Tracked by ASYNCcnt.
static __device__ inline void async_copy_tile(float* dst, const float* src,
                                              int ld, int t) {
#pragma unroll
  for (int i = 0; i < 4; ++i) {
    const int chunk = t + i * 128;           // 0..511
    const int r = chunk >> 3;                // row 0..63
    const int c = (chunk & 7) << 2;          // col in floats (16B chunks)
    // Low 32 bits of the flat shared-memory address are the LDS byte address
    // (ISA 10.2: LDS aperture keeps the offset in addr[31:0]).
    const unsigned lds = (unsigned)(uintptr_t)(dst + r * 32 + c);
    const unsigned long long ga =
        (unsigned long long)(uintptr_t)(src + (size_t)r * ld + c);
    asm volatile("global_load_async_to_lds_b128 %0, %1, off"
                 :: "v"(lds), "v"(ga) : "memory");
  }
}

static __device__ inline void wait_async_all() {
  asm volatile("s_wait_asynccnt 0x0" ::: "memory");
}

// ---------------------------------------------------------------------------
// Tiled GEMM:  C[M,N] = A[M,K] * W[N,K]^T  (+ rscale * R[M,N] if HAS_R)
// Grid: (N/64, M/64); 128 threads = 4 waves; each wave owns a 16-row strip
// and produces 4 16x16 WMMA tiles.  A/W k-slices double-buffered in LDS via
// async copies.
// ---------------------------------------------------------------------------
template <bool HAS_R>
__global__ __launch_bounds__(128) void gemm_xwt_kernel(
    const float* __restrict__ A, int lda,
    const float* __restrict__ W, int ldw,
    float* __restrict__ C, int ldc,
    const float* __restrict__ R, float rscale,
    int K) {
  __shared__ float As[2][64 * 32];   // 8 KB each buffer
  __shared__ float Ws[2][64 * 32];
  const int t = threadIdx.x;
  const int wave = t >> 5;
  const int lane = t & 31;
  const int bm = blockIdx.y << 6;
  const int bn = blockIdx.x << 6;
  const float* Ablk = A + (size_t)bm * lda;
  const float* Wblk = W + (size_t)bn * ldw;

  async_copy_tile(As[0], Ablk, lda, t);
  async_copy_tile(Ws[0], Wblk, ldw, t);
  wait_async_all();
  __syncthreads();

  v8f acc[4] = {};
  for (int kk = 0; kk < K; kk += 32) {
    const int cur = (kk >> 5) & 1;
    if (kk + 32 < K) {   // prefetch next k-slice into the other buffer
      async_copy_tile(As[cur ^ 1], Ablk + kk + 32, lda, t);
      async_copy_tile(Ws[cur ^ 1], Wblk + kk + 32, ldw, t);
    }
    const v16bf a =
        frag_row(&As[cur][(wave * 16 + (lane & 15)) * 32 + ((lane >> 4) << 4)]);
#pragma unroll
    for (int n = 0; n < 4; ++n) {
      const v16bf b =
          frag_row(&Ws[cur][(n * 16 + (lane & 15)) * 32 + ((lane >> 4) << 4)]);
      acc[n] = __builtin_amdgcn_wmma_f32_16x16x32_bf16(false, a, false, b,
                                                       (short)0, acc[n],
                                                       false, false);
    }
    wait_async_all();
    __syncthreads();
  }

  const int mrow = bm + wave * 16 + ((lane >> 4) << 3);
#pragma unroll
  for (int n = 0; n < 4; ++n) {
    const int ncol = bn + n * 16 + (lane & 15);
#pragma unroll
    for (int i = 0; i < 8; ++i) {
      float o = acc[n][i];
      if (HAS_R) o += rscale * R[(size_t)(mrow + i) * ldc + ncol];
      C[(size_t)(mrow + i) * ldc + ncol] = o;
    }
  }
}

// ---------------------------------------------------------------------------
// RMSNorm: one 256-thread block per row.  dst may alias src (same stride).
// ---------------------------------------------------------------------------
__global__ __launch_bounds__(256) void rmsnorm_kernel(
    const float* __restrict__ src, int sstride,
    const float* __restrict__ w,
    float* __restrict__ dst, int dstride, int cols) {
  __shared__ float red[256];
  const int row = blockIdx.x;
  const float* s = src + (size_t)row * sstride;
  float acc = 0.f;
  for (int c = threadIdx.x; c < cols; c += 256) { float v = s[c]; acc += v * v; }
  red[threadIdx.x] = acc;
  __syncthreads();
  for (int off = 128; off > 0; off >>= 1) {
    if (threadIdx.x < off) red[threadIdx.x] += red[threadIdx.x + off];
    __syncthreads();
  }
  const float r = rsqrtf(red[0] / (float)cols + EPS_);
  float* d = dst + (size_t)row * dstride;
  for (int c = threadIdx.x; c < cols; c += 256) d[c] = s[c] * r * w[c];
}

// ---------------------------------------------------------------------------
// RoPE + repack:  qf[NH,S,QK], k_out[NH,S,QK], v_out[NH,S,VD]
//   q   : [S, NH*QK]          (q_nope | q_pe)
//   kv  : [S, NH*(NOPE+VD)]   (k_nope | v per head)
//   ckv : [S, KVLORA+ROPE]    (c_kv | k_pe)
// Interleaved-rope: out[j]    = x[2j]*cos - x[2j+1]*sin
//                   out[32+j] = x[2j+1]*cos + x[2j]*sin
// ---------------------------------------------------------------------------
__global__ __launch_bounds__(64) void rope_pack_kernel(
    const float* __restrict__ q, const float* __restrict__ kv,
    const float* __restrict__ ckv, const int* __restrict__ pos,
    float* __restrict__ qf, float* __restrict__ kout, float* __restrict__ vout) {
  const int s = blockIdx.x, h = blockIdx.y, t = threadIdx.x;
  const float p = (float)pos[s];
  const float* qrow  = q  + (size_t)s * (NH_ * QK_) + h * QK_;
  const float* kvrow = kv + (size_t)s * (NH_ * (NOPE_ + VD_)) + h * (NOPE_ + VD_);
  float* qfrow = qf   + ((size_t)h * S_ + s) * QK_;
  float* krow  = kout + ((size_t)h * S_ + s) * QK_;
  float* vrow  = vout + ((size_t)h * S_ + s) * VD_;
  for (int d = t; d < NOPE_; d += 64) {
    qfrow[d] = qrow[d];
    krow[d]  = kvrow[d];
    vrow[d]  = kvrow[NOPE_ + d];
  }
  if (t < 32) {
    // inv_freq = 10000^(-t/32) = exp(-t * ln(10000)/32)
    const float invf = __expf(-(float)t * 0.28782313662425572f);
    const float ang = p * invf;
    const float cs = cosf(ang), sn = sinf(ang);
    const float x0 = qrow[NOPE_ + 2 * t], x1 = qrow[NOPE_ + 2 * t + 1];
    qfrow[NOPE_ + t]      = x0 * cs - x1 * sn;
    qfrow[NOPE_ + 32 + t] = x1 * cs + x0 * sn;
    const float* kpe = ckv + (size_t)s * (KVLORA_ + ROPE_) + KVLORA_;
    const float y0 = kpe[2 * t], y1 = kpe[2 * t + 1];
    krow[NOPE_ + t]      = y0 * cs - y1 * sn;
    krow[NOPE_ + 32 + t] = y1 * cs + y0 * sn;
  }
}

// ---------------------------------------------------------------------------
// Fused attention: one workgroup = (head h, 16 query rows).  4 waves.
// LDS holds the full 16 x 2048 score/prob row block.  QK^T and P*V via WMMA.
// attn out layout: [S, NH*VD] (row-major, ready for the O-projection GEMM).
// ---------------------------------------------------------------------------
__global__ __launch_bounds__(128) void attn_kernel(
    const float* __restrict__ qf, const float* __restrict__ k,
    const float* __restrict__ v, const float* __restrict__ mask,
    float* __restrict__ attn) {
  extern __shared__ float smem[];            // [16*S_] probs + reduction scratch
  float* red     = smem + 16 * S_;           // 128 partials
  float* rowstat = red + 128;                // 16 row stats

  const int h = blockIdx.y;
  const int qb = blockIdx.x * 16;
  const int wave = threadIdx.x >> 5;
  const int lane = threadIdx.x & 31;

  const float* qh = qf + ((size_t)h * S_ + qb) * QK_;
  const float* kh = k + (size_t)h * S_ * QK_;
  const float* vh = v + (size_t)h * S_ * VD_;

  // Q fragments (16 x 192 = 6 K-slices), kept in registers, reused 32x/wave.
  v16bf qa[6];
#pragma unroll
  for (int t = 0; t < 6; ++t)
    qa[t] = frag_row(qh + (size_t)(lane & 15) * QK_ + ((lane >> 4) << 4) + t * 32);

  const float scale = 0.07216878364870323f;  // 1/sqrt(192)

  // ---- Phase 1: scores = scale * Q K^T + mask -> LDS -------------------
  for (int kb = wave; kb < S_ / 16; kb += 4) {
    v8f c = {};
#pragma unroll
    for (int t = 0; t < 6; ++t) {
      v16bf b = frag_row(kh + (size_t)(kb * 16 + (lane & 15)) * QK_ +
                         ((lane >> 4) << 4) + t * 32);
      c = __builtin_amdgcn_wmma_f32_16x16x32_bf16(false, qa[t], false, b,
                                                  (short)0, c, false, false);
    }
    const int col = kb * 16 + (lane & 15);
#pragma unroll
    for (int i = 0; i < 8; ++i) {
      const int m = i + ((lane >> 4) << 3);
      smem[m * S_ + col] = c[i] * scale + mask[(size_t)(qb + m) * S_ + col];
    }
  }
  __syncthreads();

  // ---- Phase 2: row softmax over 2048 (8 threads per row) --------------
  const int row = threadIdx.x >> 3;
  const int sub = threadIdx.x & 7;
  float* srow = smem + row * S_ + sub * 256;
  float mx = -3.4e38f;
  for (int c2 = 0; c2 < 256; ++c2) mx = fmaxf(mx, srow[c2]);
  red[row * 8 + sub] = mx;
  __syncthreads();
  if (sub == 0) {
    float m2 = red[row * 8];
    for (int j = 1; j < 8; ++j) m2 = fmaxf(m2, red[row * 8 + j]);
    rowstat[row] = m2;
  }
  __syncthreads();
  const float rmax = rowstat[row];
  float sm = 0.f;
  for (int c2 = 0; c2 < 256; ++c2) {
    const float e = __expf(srow[c2] - rmax);
    srow[c2] = e;
    sm += e;
  }
  red[row * 8 + sub] = sm;
  __syncthreads();
  if (sub == 0) {
    float ssum = 0.f;
    for (int j = 0; j < 8; ++j) ssum += red[row * 8 + j];
    rowstat[row] = 1.0f / ssum;
  }
  __syncthreads();
  const float inv = rowstat[row];
  for (int c2 = 0; c2 < 256; ++c2) srow[c2] *= inv;
  __syncthreads();

  // ---- Phase 3: attn = P * V  (P from LDS, V via transposed fragments) -
  for (int ct = wave; ct < VD_ / 16; ct += 4) {
    v8f c = {};
    for (int kk = 0; kk < S_; kk += 32) {
      v16bf a = frag_row(smem + (size_t)(lane & 15) * S_ + kk + ((lane >> 4) << 4));
      v16bf b = frag_bT(vh + (size_t)kk * VD_ + ct * 16, VD_);
      c = __builtin_amdgcn_wmma_f32_16x16x32_bf16(false, a, false, b,
                                                  (short)0, c, false, false);
    }
#pragma unroll
    for (int i = 0; i < 8; ++i) {
      const int m = i + ((lane >> 4) << 3);
      attn[(size_t)(qb + m) * (NH_ * VD_) + h * VD_ + ct * 16 + (lane & 15)] = c[i];
    }
  }
}

// ---------------------------------------------------------------------------
// Launch: full pipeline.  Workspace layout (floats), with region reuse:
//   x    @ 0          (10,485,760)   -- reused later as qf (6,291,456)
//   q_a  @ 10,485,760 ( 3,145,728)   -- RMSNorm done in place
//   q    @ 13,631,488 ( 6,291,456)   -- reused later as attn (4,194,304)
//   ckv  @ 19,922,944 ( 1,179,648)
//   ckvn @ 21,102,592 ( 1,048,576)
//   kv   @ 22,151,168 ( 8,388,608)
//   total 30,539,776 floats = 122.2 MB
// d_out: hidden[2048*5120] | k[16*2048*192] | v[16*2048*128]
// ---------------------------------------------------------------------------
extern "C" void kernel_launch(void* const* d_in, const int* in_sizes, int n_in,
                              void* d_out, int out_size, void* d_ws, size_t ws_size,
                              hipStream_t stream) {
  const float* hidden    = (const float*)d_in[0];
  const int*   pos       = (const int*)  d_in[1];
  const float* mask      = (const float*)d_in[2];
  const float* ln_w      = (const float*)d_in[3];
  const float* q_a_w     = (const float*)d_in[4];
  const float* q_a_ln_w  = (const float*)d_in[5];
  const float* q_b_w     = (const float*)d_in[6];
  const float* kv_a_w    = (const float*)d_in[7];
  const float* kv_a_ln_w = (const float*)d_in[8];
  const float* kv_b_w    = (const float*)d_in[9];
  const float* o_w       = (const float*)d_in[10];

  float* ws   = (float*)d_ws;
  float* x    = ws;                    // S x H normed hidden
  float* qf   = ws;                    // [NH,S,QK]  (reuses x region)
  float* q_a  = ws + 10485760;         // S x QLORA
  float* q    = ws + 13631488;         // S x NH*QK
  float* attn = q;                     // S x NH*VD  (reuses q region)
  float* ckv  = ws + 19922944;         // S x (KVLORA+ROPE)
  float* ckvn = ws + 21102592;         // S x KVLORA
  float* kv   = ws + 22151168;         // S x NH*(NOPE+VD)

  float* out_hidden = (float*)d_out;
  float* out_k = out_hidden + (size_t)S_ * H_;                 // [NH,S,QK]
  float* out_v = out_k + (size_t)NH_ * S_ * QK_;               // [NH,S,VD]

  // 1. x = rms_norm(hidden, ln_w)
  rmsnorm_kernel<<<S_, 256, 0, stream>>>(hidden, H_, ln_w, x, H_, H_);
  // 2. q_a = x @ q_a_w^T
  gemm_xwt_kernel<false><<<dim3(QLORA_ / 64, S_ / 64), 128, 0, stream>>>(
      x, H_, q_a_w, H_, q_a, QLORA_, nullptr, 0.f, H_);
  // 3. q_a = rms_norm(q_a, q_a_ln_w)   (in place)
  rmsnorm_kernel<<<S_, 256, 0, stream>>>(q_a, QLORA_, q_a_ln_w, q_a, QLORA_, QLORA_);
  // 4. q = q_a @ q_b_w^T
  gemm_xwt_kernel<false><<<dim3((NH_ * QK_) / 64, S_ / 64), 128, 0, stream>>>(
      q_a, QLORA_, q_b_w, QLORA_, q, NH_ * QK_, nullptr, 0.f, QLORA_);
  // 5. ckv = x @ kv_a_w^T
  gemm_xwt_kernel<false><<<dim3((KVLORA_ + ROPE_) / 64, S_ / 64), 128, 0, stream>>>(
      x, H_, kv_a_w, H_, ckv, KVLORA_ + ROPE_, nullptr, 0.f, H_);
  // 6. ckvn = rms_norm(ckv[:, :512], kv_a_ln_w)
  rmsnorm_kernel<<<S_, 256, 0, stream>>>(ckv, KVLORA_ + ROPE_, kv_a_ln_w,
                                         ckvn, KVLORA_, KVLORA_);
  // 7. kv = ckvn @ kv_b_w^T
  gemm_xwt_kernel<false><<<dim3((NH_ * (NOPE_ + VD_)) / 64, S_ / 64), 128, 0, stream>>>(
      ckvn, KVLORA_, kv_b_w, KVLORA_, kv, NH_ * (NOPE_ + VD_), nullptr, 0.f, KVLORA_);
  // 8. RoPE + repack -> qf, k (d_out), v (d_out)
  rope_pack_kernel<<<dim3(S_, NH_), 64, 0, stream>>>(q, kv, ckv, pos, qf, out_k, out_v);
  // 9. fused attention -> attn [S, NH*VD]
  const size_t smemBytes = (size_t)(16 * S_ + 128 + 16) * sizeof(float);
  attn_kernel<<<dim3(S_ / 16, NH_), 128, smemBytes, stream>>>(qf, out_k, out_v, mask, attn);
  // 10. hidden_out = hidden/8 + attn @ o_w^T
  gemm_xwt_kernel<true><<<dim3(H_ / 64, S_ / 64), 128, 0, stream>>>(
      attn, NH_ * VD_, o_w, NH_ * VD_, out_hidden, H_, hidden, 0.125f, NH_ * VD_);
}
